// ExtendedS5SSM_85907935855371
// MI455X (gfx1250) — matrix operand order repaired and checked by reading, not compiled, not run on hardware
//
#include <hip/hip_runtime.h>

// Extended S5 SSM for MI455X (gfx1250, wave32, WMMA).
// All large GEMMs: v_wmma_f32_16x16x32_bf16 (bf16 operands converted on the
// fly with v_cvt_pk_bf16_f32, f32 accumulate), double-buffered LDS pipeline,
// compile-time shapes so stores/loads use immediate offsets. Sequential
// recurrence: 3-phase chunked scan in full f32 VALU.

#define L_SEQ 16384
#define H_DIM 1024
#define P_DIM 1024
#define R_DIM 512
#define CHUNK 128
#define NCHUNK (L_SEQ / CHUNK)

typedef __attribute__((ext_vector_type(16))) __bf16 v16bf;
typedef __attribute__((ext_vector_type(4)))  __bf16 v4bf;
typedef __attribute__((ext_vector_type(8)))  float  v8f;
typedef __attribute__((ext_vector_type(4)))  float  v4f;

union Frag16 {
    v16bf v;
    uint4 q[2];
};

union PKH {
    v4bf v;
    uint2 u;
};

// 4x f32 -> 4x bf16 (hardware v_cvt_pk_bf16_f32), packed into 8 bytes
__device__ __forceinline__ uint2 cvt4(const float4 f, bool neg) {
    v4f x = {f.x, f.y, f.z, f.w};
    if (neg) x = -x;
    PKH p;
    p.v = __builtin_convertvector(x, v4bf);
    return p.u;
}

// 8x f32 -> 8x bf16 packed into 16 bytes (one ds_store_b128)
__device__ __forceinline__ uint4 cvt8(const float4 lo, const float4 hi, bool neg) {
    const uint2 a = cvt4(lo, neg);
    const uint2 b = cvt4(hi, neg);
    return uint4{a.x, a.y, b.x, b.y};
}

// ---------------------------------------------------------------------------
// Tiled GEMM: C[M,N] (= or +=) A[M,K] @ B[N,K]^T, f32 in memory, bf16 WMMA.
// N, K compile-time; M = L_SEQ. 128x128 block tile, BK=32, 8 waves (2x4),
// each wave 64x32 = 4x2 WMMA accumulators. Double-buffered LDS, one barrier
// per K-step; next tile staged in registers while WMMAs run.
// ---------------------------------------------------------------------------
#define BM 128
#define BN 128
#define BK 32
#define LDSK 40   // bf16 elems per LDS row (32 + 8 pad); keeps 16B alignment

template <int N, int K, bool ACC, bool NEG>
__global__ __launch_bounds__(256) void gemm_bf16(const float* __restrict__ A,
                                                 const float* __restrict__ B,
                                                 float* __restrict__ C) {
    __shared__ __align__(16) unsigned short sA[2][BM * LDSK];
    __shared__ __align__(16) unsigned short sB[2][BN * LDSK];
    constexpr int STEPS = K / BK;

    const int tid     = threadIdx.x;
    const int lane    = tid & 31;
    const int wave    = tid >> 5;      // 8 waves
    const int waveM   = wave >> 2;     // 0..1 -> 64-row slab
    const int waveN   = wave & 3;      // 0..3 -> 32-col slab
    const int lane15  = lane & 15;
    const int halfsel = lane >> 4;     // K-half select per ISA frag layout
    const int blockN0 = blockIdx.x * BN;
    const int blockM0 = blockIdx.y * BM;

    // cooperative load mapping: thread owns 8 contiguous cols in rows
    // {row0, row0+64} of both tiles -> 2x b128 LDS store per matrix.
    const int row0 = tid >> 2;          // 0..63
    const int col8 = (tid & 3) << 3;    // 0,8,16,24

    const float* pA0 = A + (size_t)(blockM0 + row0) * K + col8;
    const float* pA1 = pA0 + (size_t)64 * K;
    const float* pB0 = B + (size_t)(blockN0 + row0) * K + col8;
    const float* pB1 = pB0 + (size_t)64 * K;

    const int ldsOff0 = row0 * LDSK + col8;
    const int ldsOff1 = (row0 + 64) * LDSK + col8;

    v8f acc[4][2] = {};

    // ---- preload tile 0 ----
    {
        const float4 a00 = *(const float4*)pA0;
        const float4 a01 = *(const float4*)(pA0 + 4);
        const float4 a10 = *(const float4*)pA1;
        const float4 a11 = *(const float4*)(pA1 + 4);
        const float4 b00 = *(const float4*)pB0;
        const float4 b01 = *(const float4*)(pB0 + 4);
        const float4 b10 = *(const float4*)pB1;
        const float4 b11 = *(const float4*)(pB1 + 4);
        *(uint4*)(&sA[0][ldsOff0]) = cvt8(a00, a01, NEG);
        *(uint4*)(&sA[0][ldsOff1]) = cvt8(a10, a11, NEG);
        *(uint4*)(&sB[0][ldsOff0]) = cvt8(b00, b01, false);
        *(uint4*)(&sB[0][ldsOff1]) = cvt8(b10, b11, false);
    }
    __syncthreads();

#pragma unroll 2
    for (int ks = 0; ks < STEPS; ++ks) {
        const int cur = ks & 1;

        // stage next tile's global loads (in flight during WMMAs)
        float4 a00, a01, a10, a11, b00, b01, b10, b11;
        if (ks + 1 < STEPS) {
            pA0 += BK; pA1 += BK; pB0 += BK; pB1 += BK;
            a00 = *(const float4*)pA0;
            a01 = *(const float4*)(pA0 + 4);
            a10 = *(const float4*)pA1;
            a11 = *(const float4*)(pA1 + 4);
            b00 = *(const float4*)pB0;
            b01 = *(const float4*)(pB0 + 4);
            b10 = *(const float4*)pB1;
            b11 = *(const float4*)(pB1 + 4);
        }
        // prefetch two steps ahead into cache (global_prefetch_b8)
        if (ks + 2 < STEPS) {
            __builtin_prefetch(pA0 + BK, 0, 1);
            __builtin_prefetch(pA1 + BK, 0, 1);
            __builtin_prefetch(pB0 + BK, 0, 1);
            __builtin_prefetch(pB1 + BK, 0, 1);
        }

        // A fragment (16x32 bf16): lane l holds M=l&15.
        //   lanes 0-15 : e0-7 -> K 0-7,  e8-15 -> K 16-23
        //   lanes 16-31: e0-7 -> K 8-15, e8-15 -> K 24-31
        Frag16 afrag[4];
#pragma unroll
        for (int mt = 0; mt < 4; ++mt) {
            const int r  = waveM * 64 + mt * 16 + lane15;
            const int kb = halfsel * 8;
            afrag[mt].q[0] = *(const uint4*)(&sA[cur][r * LDSK + kb]);
            afrag[mt].q[1] = *(const uint4*)(&sA[cur][r * LDSK + kb + 16]);
        }
        // B fragment (32x16 bf16): lane l holds N=l&15; same K order as A:
        // lanes 0-15 K 0-7,16-23; lanes 16-31 K 8-15,24-31.
        Frag16 bfrag[2];
#pragma unroll
        for (int nt = 0; nt < 2; ++nt) {
            const int cb = waveN * 32 + nt * 16 + lane15;
            const int kb = halfsel * 8;
            bfrag[nt].q[0] = *(const uint4*)(&sB[cur][cb * LDSK + kb]);
            bfrag[nt].q[1] = *(const uint4*)(&sB[cur][cb * LDSK + kb + 16]);
        }

#pragma unroll
        for (int mt = 0; mt < 4; ++mt)
#pragma unroll
            for (int nt = 0; nt < 2; ++nt)
                acc[mt][nt] = __builtin_amdgcn_wmma_f32_16x16x32_bf16(
                    false, afrag[mt].v, false, bfrag[nt].v, (short)0,
                    acc[mt][nt], false, false);

        // convert + store next tile into the other buffer
        if (ks + 1 < STEPS) {
            const int nxt = cur ^ 1;
            *(uint4*)(&sA[nxt][ldsOff0]) = cvt8(a00, a01, NEG);
            *(uint4*)(&sA[nxt][ldsOff1]) = cvt8(a10, a11, NEG);
            *(uint4*)(&sB[nxt][ldsOff0]) = cvt8(b00, b01, false);
            *(uint4*)(&sB[nxt][ldsOff1]) = cvt8(b10, b11, false);
        }
        __syncthreads();
    }

    // Epilogue. C/D layout: VGPR r, lanes 0-15 -> (M=r, N=lane),
    // lanes 16-31 -> (M=r+8, N=lane-16). With N constexpr all offsets are
    // immediates off a single per-lane base.
    float* cb = C + (size_t)(blockM0 + waveM * 64 + halfsel * 8) * N +
                blockN0 + waveN * 32 + lane15;
#pragma unroll
    for (int mt = 0; mt < 4; ++mt) {
#pragma unroll
        for (int nt = 0; nt < 2; ++nt) {
#pragma unroll
            for (int r = 0; r < 8; ++r) {
                const int off = (mt * 16 + r) * N + nt * 16;
                const float v = acc[mt][nt][r];
                if (ACC) cb[off] += v; else cb[off] = v;
            }
        }
    }
}

// ---------------------------------------------------------------------------
// Chunked diagonal complex scan: x[t] = lambda*x[t-1] + b[t], in place.
// Layout (L, P) row-major; thread = (chunk c, channel p), coalesced over p.
// ---------------------------------------------------------------------------
__global__ __launch_bounds__(256) void scan_local(float* __restrict__ xr,
                                                  float* __restrict__ xi,
                                                  const float* __restrict__ lre,
                                                  const float* __restrict__ lim,
                                                  float* __restrict__ cre,
                                                  float* __restrict__ cim) {
    const int p = blockIdx.x * 256 + threadIdx.x;
    const int c = blockIdx.y;
    const float lr = lre[p], li = lim[p];
    float sr = 0.f, si = 0.f;
    size_t idx = (size_t)c * CHUNK * P_DIM + p;
    for (int i = 0; i < CHUNK; ++i) {
        const float br = xr[idx], bi = xi[idx];
        const float nr = fmaf(lr, sr, fmaf(-li, si, br));
        const float ni = fmaf(lr, si, fmaf(li, sr, bi));
        xr[idx] = nr;
        xi[idx] = ni;
        sr = nr;
        si = ni;
        idx += P_DIM;
    }
    cre[(size_t)c * P_DIM + p] = sr;
    cim[(size_t)c * P_DIM + p] = si;
}

__global__ __launch_bounds__(256) void scan_carry(const float* __restrict__ lre,
                                                  const float* __restrict__ lim,
                                                  const float* __restrict__ cre,
                                                  const float* __restrict__ cim,
                                                  float* __restrict__ pre,
                                                  float* __restrict__ pim) {
    const int p = blockIdx.x * 256 + threadIdx.x;
    const float lr = lre[p], li = lim[p];
    // lambda^CHUNK
    float ar = 1.f, ai = 0.f;
    for (int i = 0; i < CHUNK; ++i) {
        const float nr = ar * lr - ai * li;
        const float ni = ar * li + ai * lr;
        ar = nr;
        ai = ni;
    }
    float sr = 0.f, si = 0.f;
    for (int c = 0; c < NCHUNK; ++c) {
        const size_t o = (size_t)c * P_DIM + p;
        pre[o] = sr;  // exclusive prefix: state entering chunk c
        pim[o] = si;
        const float nr = ar * sr - ai * si + cre[o];
        const float ni = ar * si + ai * sr + cim[o];
        sr = nr;
        si = ni;
    }
}

__global__ __launch_bounds__(256) void scan_apply(float* __restrict__ xr,
                                                  float* __restrict__ xi,
                                                  const float* __restrict__ lre,
                                                  const float* __restrict__ lim,
                                                  const float* __restrict__ pre,
                                                  const float* __restrict__ pim) {
    const int p = blockIdx.x * 256 + threadIdx.x;
    const int c = blockIdx.y;
    const float lr = lre[p], li = lim[p];
    const float sr = pre[(size_t)c * P_DIM + p];
    const float si = pim[(size_t)c * P_DIM + p];
    float cr = lr * sr - li * si;  // lambda^1 * s_in
    float ci = lr * si + li * sr;
    size_t idx = (size_t)c * CHUNK * P_DIM + p;
    for (int i = 0; i < CHUNK; ++i) {
        xr[idx] += cr;
        xi[idx] += ci;
        const float nr = cr * lr - ci * li;
        const float ni = cr * li + ci * lr;
        cr = nr;
        ci = ni;
        idx += P_DIM;
    }
}

// ---------------------------------------------------------------------------
// Small helpers
// ---------------------------------------------------------------------------
// MT[r,p] = sum_{r2} F[r2,p] * Delta[r2,r]   (so p_seq = x @ MT^T = x@F^T@Delta)
__global__ __launch_bounds__(256) void build_mt(const float* __restrict__ F,
                                                const float* __restrict__ Delta,
                                                float* __restrict__ MT) {
    const int idx = blockIdx.x * 256 + threadIdx.x;
    const int r = idx / P_DIM;
    const int p = idx - r * P_DIM;
    float acc = 0.f;
    for (int r2 = 0; r2 < R_DIM; ++r2)
        acc = fmaf(F[(size_t)r2 * P_DIM + p], Delta[(size_t)r2 * R_DIM + r], acc);
    MT[idx] = acc;
}

// u_shift[t] = u[t+1] (zeros in last row)
__global__ __launch_bounds__(256) void shift_u(const float* __restrict__ u,
                                               float* __restrict__ us) {
    const size_t idx = (size_t)blockIdx.x * 256 + threadIdx.x;
    const size_t t = idx / H_DIM;
    us[idx] = (t + 1 < L_SEQ) ? u[idx + H_DIM] : 0.f;
}

// out[0]=0 ; out[t] = Y[t-1] + D*u[t]
__global__ __launch_bounds__(256) void finalize(const float* __restrict__ Y,
                                                const float* __restrict__ u,
                                                const float* __restrict__ D,
                                                float* __restrict__ out) {
    const size_t idx = (size_t)blockIdx.x * 256 + threadIdx.x;
    const size_t t = idx / H_DIM;
    const int h = (int)(idx - t * H_DIM);
    out[idx] = (t == 0) ? 0.f : (Y[idx - H_DIM] + D[h] * u[idx]);
}

// ---------------------------------------------------------------------------
extern "C" void kernel_launch(void* const* d_in, const int* in_sizes, int n_in,
                              void* d_out, int out_size, void* d_ws,
                              size_t ws_size, hipStream_t stream) {
    (void)in_sizes; (void)n_in; (void)out_size; (void)ws_size;
    const float* u     = (const float*)d_in[0];
    const float* l_re  = (const float*)d_in[1];
    const float* l_im  = (const float*)d_in[2];
    const float* B_re  = (const float*)d_in[3];
    const float* B_im  = (const float*)d_in[4];
    const float* C_re  = (const float*)d_in[5];
    const float* C_im  = (const float*)d_in[6];
    const float* Dv    = (const float*)d_in[7];
    const float* F     = (const float*)d_in[8];
    const float* E     = (const float*)d_in[9];
    const float* Delta = (const float*)d_in[10];
    float* out = (float*)d_out;

    char* ws = (char*)d_ws;
    const size_t LP = (size_t)L_SEQ * P_DIM * sizeof(float);  // 64 MB
    const size_t LR = (size_t)L_SEQ * R_DIM * sizeof(float);  // 32 MB
    const size_t PC = (size_t)NCHUNK * P_DIM;                 // carry elems
    float* bufA = (float*)(ws);             // Bu1_re -> x1_re -> Bu2_re -> x2_re
    float* bufB = (float*)(ws + LP);        // Bu1_im -> x1_im -> Bu2_im -> x2_im
    float* bufC = (float*)(ws + 2 * LP);    // u_shift -> Y (x_aug)
    float* pseq = (float*)(ws + 3 * LP);    // (L, R)
    float* cre  = (float*)(ws + 3 * LP + LR);
    float* cim  = cre + PC;
    float* pre  = cim + PC;
    float* pim  = pre + PC;
    float* MT   = pim + PC;                 // (R, P)

    const dim3 blk(256);
    const dim3 gP(P_DIM / BN, L_SEQ / BM);
    const dim3 gR(R_DIM / BN, L_SEQ / BM);
    const dim3 gH(H_DIM / BN, L_SEQ / BM);
    const dim3 sg(P_DIM / 256, NCHUNK);
    const dim3 cg(P_DIM / 256);
    const dim3 eg((unsigned)(((size_t)L_SEQ * H_DIM) / 256));

    // 1) Bu1 = u @ B^T (complex)
    gemm_bf16<P_DIM, H_DIM, false, false><<<gP, blk, 0, stream>>>(u, B_re, bufA);
    gemm_bf16<P_DIM, H_DIM, false, false><<<gP, blk, 0, stream>>>(u, B_im, bufB);
    // 2) scan 1 (in place -> x1)
    scan_local<<<sg, blk, 0, stream>>>(bufA, bufB, l_re, l_im, cre, cim);
    scan_carry<<<cg, blk, 0, stream>>>(l_re, l_im, cre, cim, pre, pim);
    scan_apply<<<sg, blk, 0, stream>>>(bufA, bufB, l_re, l_im, pre, pim);
    // 3) p_seq = x1_re @ (F^T Delta)   via MT = Delta^T F
    build_mt<<<dim3((R_DIM * P_DIM) / 256), blk, 0, stream>>>(F, Delta, MT);
    gemm_bf16<R_DIM, P_DIM, false, false><<<gR, blk, 0, stream>>>(bufA, MT, pseq);
    // 4) u_shift
    shift_u<<<eg, blk, 0, stream>>>(u, bufC);
    // 5) Bu2_re = u_shift @ B_re^T + p_seq @ E^T ;  Bu2_im = u_shift @ B_im^T
    gemm_bf16<P_DIM, H_DIM, false, false><<<gP, blk, 0, stream>>>(bufC, B_re, bufA);
    gemm_bf16<P_DIM, R_DIM, true,  false><<<gP, blk, 0, stream>>>(pseq, E, bufA);
    gemm_bf16<P_DIM, H_DIM, false, false><<<gP, blk, 0, stream>>>(bufC, B_im, bufB);
    // 6) scan 2 (in place -> x2)
    scan_local<<<sg, blk, 0, stream>>>(bufA, bufB, l_re, l_im, cre, cim);
    scan_carry<<<cg, blk, 0, stream>>>(l_re, l_im, cre, cim, pre, pim);
    scan_apply<<<sg, blk, 0, stream>>>(bufA, bufB, l_re, l_im, pre, pim);
    // 7) x_aug = x2_re @ C_re^T - x2_im @ C_im^T  (into bufC; u_shift dead)
    gemm_bf16<H_DIM, P_DIM, false, false><<<gH, blk, 0, stream>>>(bufA, C_re, bufC);
    gemm_bf16<H_DIM, P_DIM, true,  true ><<<gH, blk, 0, stream>>>(bufB, C_im, bufC);
    // 8) output = shift(x_aug) + D*u
    finalize<<<eg, blk, 0, stream>>>(bufC, u, Dv, out);
}
